// ModulatedSparseTransformerBlock_14568529068665
// MI455X (gfx1250) — compile-verified, matrix operand order checked by reading, not run on hardware
//
#include <hip/hip_runtime.h>
#include <hip/hip_bf16.h>
#include <cmath>

// ---------------------------------------------------------------------------
// Types for CDNA5 WMMA (wave32): v_wmma_f32_16x16x32_bf16, TDM descriptors
// ---------------------------------------------------------------------------
typedef __bf16 bf16_t;
typedef __attribute__((ext_vector_type(16))) __bf16 v16bf;
typedef __attribute__((ext_vector_type(8)))  __bf16 v8bf;
typedef __attribute__((ext_vector_type(8)))  float  v8f;
typedef __attribute__((ext_vector_type(4)))  unsigned int v4u;
typedef __attribute__((ext_vector_type(8)))  int v8i_t;
typedef __attribute__((ext_vector_type(4)))  int v4i_t;

#define C_DIM 1024
#define L_DIM 2048
#define H_DIM 16
#define D_DIM 64

__device__ __forceinline__ v8f wmma_bf16(v16bf a, v16bf b, v8f c) {
  return __builtin_amdgcn_wmma_f32_16x16x32_bf16(false, a, false, b, (short)0, c,
                                                 false, false);
}

// Load a 16x32 bf16 A/B fragment from a row-major LDS tile (stride elements).
__device__ __forceinline__ v16bf load_frag(const bf16_t* base, int stride,
                                           int row0, int k0, int lane) {
  const int r  = lane & 15;
  const int lh = lane >> 4;
  const bf16_t* p = base + (row0 + r) * stride + k0 + 8 * lh;
  v16bf out;
#pragma unroll
  for (int i = 0; i < 8; ++i) out[i] = p[i];
#pragma unroll
  for (int i = 0; i < 8; ++i) out[8 + i] = p[16 + i];
  return out;
}

// ---------------------------------------------------------------------------
// TDM: issue a 2-D tensor_load_to_lds of a (rows x 32) bf16 tile.
// Source row stride = stride_elems (bf16 elems). Destination LDS rows are
// padded to 40 elems via D# pad: pad_interval code 3 (16 DWORDs = 64B = one
// 32-elem bf16 row), pad_amount code 3 (4 DWORDs = 16B = 8 bf16).
// ---------------------------------------------------------------------------
__device__ __forceinline__ void tdm_load_tile(unsigned int lds_off,
                                              const bf16_t* gptr,
                                              unsigned int rows,
                                              unsigned int tensor_d0,
                                              unsigned long long stride_elems) {
  const unsigned long long ga = (unsigned long long)(uintptr_t)gptr;
  v4u g0;
  g0[0] = 1u;                                   // count=1, user descriptor
  g0[1] = lds_off;                              // lds_addr (bytes)
  g0[2] = (unsigned int)(ga & 0xffffffffull);   // global_addr[31:0]
  g0[3] = (unsigned int)((ga >> 32) & 0x1ffffffull) | (2u << 30);  // type=2
  v8i_t g1;
  // data_size=1 (2B), pad_enable, pad_interval=3 (16DW), pad_amount=3 (4DW)
  g1[0] = (int)((1u << 16) | (1u << 20) | (3u << 22) | (3u << 25));
  const unsigned int td0 = tensor_d0;           // tensor dim0 (elements)
  const unsigned int td1 = rows;                // tensor dim1
  const unsigned int tile0 = 32u, tile1 = rows; // tile dims
  const unsigned long long s0 = stride_elems;   // dim0 stride (elements)
  g1[1] = (int)((td0 & 0xffffu) << 16);                         // bits 63:48
  g1[2] = (int)(((td0 >> 16) & 0xffffu) | ((td1 & 0xffffu) << 16));
  g1[3] = (int)(((td1 >> 16) & 0xffffu) | (tile0 << 16));
  g1[4] = (int)(tile1 & 0xffffu);                               // tile_dim1
  g1[5] = (int)(unsigned int)(s0 & 0xffffffffull);
  g1[6] = (int)((s0 >> 32) & 0xffffull);
  g1[7] = 0;
  const v4i_t z4 = {0, 0, 0, 0};
  const v8i_t z8 = {0, 0, 0, 0, 0, 0, 0, 0};
  __builtin_amdgcn_tensor_load_to_lds(g0, g1, z4, z4, z8, 0);
}

// ---------------------------------------------------------------------------
// Kernel 0: m = silu(mod) @ w_mod + b_mod   -> mod6 [2, 6144] (fp32)
// ---------------------------------------------------------------------------
__global__ void mod_silu_gemm(const float* __restrict__ mod,
                              const float* __restrict__ w_mod,
                              const float* __restrict__ b_mod,
                              float* __restrict__ mod6) {
  __shared__ float sm[C_DIM];
  const int b   = blockIdx.y;
  const int tid = threadIdx.x;
  for (int i = tid; i < C_DIM; i += 256) {
    float x = mod[b * C_DIM + i];
    sm[i] = x / (1.0f + __expf(-x));   // SiLU
  }
  __syncthreads();
  const int col = blockIdx.x * 256 + tid;
  float acc = 0.f;
  for (int k = 0; k < C_DIM; ++k)
    acc = fmaf(sm[k], w_mod[k * (6 * C_DIM) + col], acc);
  mod6[b * 6 * C_DIM + col] = acc + b_mod[col];
}

// ---------------------------------------------------------------------------
// Weight prep: Wt[N,K] bf16 = (bf16)W[K,N]  (32x32 tiles via LDS)
// ---------------------------------------------------------------------------
__global__ void transpose_cvt(const float* __restrict__ W,
                              bf16_t* __restrict__ Wt, int K, int N) {
  __shared__ float t[32][33];
  const int k0 = blockIdx.x * 32, n0 = blockIdx.y * 32;
  const int tx = threadIdx.x, ty = threadIdx.y;  // 32 x 8
#pragma unroll
  for (int i = 0; i < 4; ++i)
    t[ty + 8 * i][tx] = W[(k0 + ty + 8 * i) * N + n0 + tx];
  __syncthreads();
#pragma unroll
  for (int i = 0; i < 4; ++i)
    Wt[(n0 + ty + 8 * i) * K + k0 + tx] = (bf16_t)t[tx][ty + 8 * i];
}

// ---------------------------------------------------------------------------
// Kernel 1: h = LN(x) * (1 + sc) + sh  -> bf16   (one block per token row)
// ---------------------------------------------------------------------------
__global__ void ln_modulate(const float* __restrict__ x,
                            const float* __restrict__ mod6,
                            int sh_off, int sc_off,
                            bf16_t* __restrict__ out) {
  __shared__ float red[256];
  const int row = blockIdx.x;
  const int tid = threadIdx.x;
  const float* xr = x + row * C_DIM;
  float v0 = xr[tid], v1 = xr[tid + 256], v2 = xr[tid + 512], v3 = xr[tid + 768];
  red[tid] = v0 + v1 + v2 + v3;
  __syncthreads();
  for (int o = 128; o > 0; o >>= 1) {
    if (tid < o) red[tid] += red[tid + o];
    __syncthreads();
  }
  const float mean = red[0] * (1.0f / C_DIM);
  __syncthreads();
  const float d0 = v0 - mean, d1 = v1 - mean, d2 = v2 - mean, d3 = v3 - mean;
  red[tid] = d0 * d0 + d1 * d1 + d2 * d2 + d3 * d3;
  __syncthreads();
  for (int o = 128; o > 0; o >>= 1) {
    if (tid < o) red[tid] += red[tid + o];
    __syncthreads();
  }
  const float rstd = rsqrtf(red[0] * (1.0f / C_DIM) + 1e-6f);
  const int b = row >> 11;  // 2048 tokens per batch
  const float* shp = mod6 + b * 6 * C_DIM + sh_off;
  const float* scp = mod6 + b * 6 * C_DIM + sc_off;
  bf16_t* orow = out + row * C_DIM;
  int n;
  n = tid;       orow[n] = (bf16_t)(d0 * rstd * (1.f + scp[n]) + shp[n]);
  n = tid + 256; orow[n] = (bf16_t)(d1 * rstd * (1.f + scp[n]) + shp[n]);
  n = tid + 512; orow[n] = (bf16_t)(d2 * rstd * (1.f + scp[n]) + shp[n]);
  n = tid + 768; orow[n] = (bf16_t)(d3 * rstd * (1.f + scp[n]) + shp[n]);
}

// ---------------------------------------------------------------------------
// WMMA GEMM: out = epilogue(A_bf16[M,K] @ Bt_bf16[N,K]^T + bias)
// Block: 128 threads (4 waves). Block tile 128(M) x 64(N); wave tile 32x64.
// Both LDS tiles staged by the Tensor Data Mover, double-buffered.
// ---------------------------------------------------------------------------
enum { EPI_BF16 = 0, EPI_GELU = 1, EPI_RESID = 2 };

template <int EPI>
__global__ void gemm_wmma(const bf16_t* __restrict__ A,
                          const bf16_t* __restrict__ Bt,
                          const float* __restrict__ bias,
                          bf16_t* __restrict__ out_bf,
                          float* __restrict__ out_f,
                          const float* __restrict__ base,
                          const float* __restrict__ gate,
                          int M, int N, int K) {
  __shared__ bf16_t As[2][128 * 40];
  __shared__ bf16_t Bs[2][64 * 40];
  const int tid  = threadIdx.x;
  const int lane = tid & 31;
  const int wave = tid >> 5;
  const int m0 = blockIdx.y * 128, n0 = blockIdx.x * 64;
  const int ntiles = K >> 5;

  const unsigned int asOff0 = (unsigned int)(uintptr_t)&As[0][0];
  const unsigned int asOff1 = (unsigned int)(uintptr_t)&As[1][0];
  const unsigned int bsOff0 = (unsigned int)(uintptr_t)&Bs[0][0];
  const unsigned int bsOff1 = (unsigned int)(uintptr_t)&Bs[1][0];

  v8f acc[2][4] = {};

  if (tid < 32) {  // wave 0 drives the TDM
    tdm_load_tile(asOff0, A + (long)m0 * K, 128u, (unsigned)K, (unsigned)K);
    tdm_load_tile(bsOff0, Bt + (long)n0 * K, 64u, (unsigned)K, (unsigned)K);
  }

  for (int it = 0; it < ntiles; ++it) {
    const int cur = it & 1;
    if (tid < 32) {
      if (it + 1 < ntiles) {
        const int k1 = (it + 1) << 5;
        tdm_load_tile(cur ? asOff0 : asOff1, A + (long)m0 * K + k1, 128u,
                      (unsigned)K, (unsigned)K);
        tdm_load_tile(cur ? bsOff0 : bsOff1, Bt + (long)n0 * K + k1, 64u,
                      (unsigned)K, (unsigned)K);
        __builtin_amdgcn_s_wait_tensorcnt(2);  // current pair complete
      } else {
        __builtin_amdgcn_s_wait_tensorcnt(0);
      }
    }
    __syncthreads();

    const bf16_t* a  = &As[cur][0];
    const bf16_t* bs = &Bs[cur][0];
    v16bf af[2], bfv[4];
#pragma unroll
    for (int i = 0; i < 2; ++i)
      af[i] = load_frag(a, 40, 32 * wave + 16 * i, 0, lane);
#pragma unroll
    for (int j = 0; j < 4; ++j)
      bfv[j] = load_frag(bs, 40, 16 * j, 0, lane);
#pragma unroll
    for (int i = 0; i < 2; ++i)
#pragma unroll
      for (int j = 0; j < 4; ++j)
        acc[i][j] = wmma_bf16(af[i], bfv[j], acc[i][j]);
    __syncthreads();
  }

  const int lh = lane >> 4, ln = lane & 15;
#pragma unroll
  for (int i = 0; i < 2; ++i)
#pragma unroll
    for (int j = 0; j < 4; ++j) {
      const int n = n0 + 16 * j + ln;
      const float bv = bias[n];
#pragma unroll
      for (int e = 0; e < 8; ++e) {
        const int m = m0 + 32 * wave + 16 * i + e + 8 * lh;
        const float v = acc[i][j][e] + bv;
        if (EPI == EPI_BF16) {
          out_bf[m * N + n] = (bf16_t)v;
        } else if (EPI == EPI_GELU) {
          out_bf[m * N + n] = (bf16_t)(0.5f * v * (1.0f + erff(v * 0.70710678f)));
        } else {  // x + gate * (acc + bias)
          out_f[m * N + n] =
              base[m * N + n] + gate[(m >> 11) * (6 * C_DIM) + n] * v;
        }
      }
    }
}

// ---------------------------------------------------------------------------
// Flash attention: one block per (q-tile of 64 rows, batch*head).
// 128 threads = 4 waves; each wave owns 16 q rows. D = 64, kv tiles of 64.
// ---------------------------------------------------------------------------
__global__ void flash_attn(const bf16_t* __restrict__ qkv,
                           bf16_t* __restrict__ o) {
  __shared__ bf16_t Qt[64 * 72];
  __shared__ bf16_t Kt[64 * 72];
  __shared__ bf16_t Vt[64 * 72];  // transposed: [d][kv]
  __shared__ float  Ss[64 * 65];
  __shared__ bf16_t Ps[64 * 72];
  __shared__ float mrow[64], lrow[64], arow[64];

  const int tid  = threadIdx.x;
  const int lane = tid & 31;
  const int wave = tid >> 5;
  const int lh = lane >> 4, ln = lane & 15;
  const int qt = blockIdx.x, bh = blockIdx.y;
  const int b = bh >> 4, head = bh & 15;
  const int tok0 = b * L_DIM + qt * 64;
  const int cq   = head * D_DIM;

  {  // stage Q tile
    const int row = tid >> 1, half = tid & 1;
    const bf16_t* src = qkv + (tok0 + row) * (3 * C_DIM) + cq + 32 * half;
#pragma unroll
    for (int i = 0; i < 32; ++i) Qt[row * 72 + 32 * half + i] = src[i];
  }
  if (tid < 64) { mrow[tid] = -1e30f; lrow[tid] = 0.f; }
  v8f oacc[4] = {};
  __syncthreads();

  for (int t = 0; t < L_DIM / 64; ++t) {
    const int kv0 = b * L_DIM + t * 64;
    {  // stage K (row-major) and V (transposed)
      const int row = tid >> 1, half = tid & 1;
      const bf16_t* ks = qkv + (kv0 + row) * (3 * C_DIM) + C_DIM + cq + 32 * half;
#pragma unroll
      for (int i = 0; i < 32; ++i) Kt[row * 72 + 32 * half + i] = ks[i];
      const bf16_t* vs = qkv + (kv0 + row) * (3 * C_DIM) + 2 * C_DIM + cq + 32 * half;
#pragma unroll
      for (int i = 0; i < 32; ++i) Vt[(32 * half + i) * 72 + row] = vs[i];
    }
    __syncthreads();

    // S = (Q K^T) * 1/sqrt(64)
    v8f sacc[4] = {};
#pragma unroll
    for (int ks = 0; ks < 64; ks += 32) {
      v16bf aq = load_frag(Qt, 72, 16 * wave, ks, lane);
#pragma unroll
      for (int f = 0; f < 4; ++f) {
        v16bf bk = load_frag(Kt, 72, 16 * f, ks, lane);
        sacc[f] = wmma_bf16(aq, bk, sacc[f]);
      }
    }
#pragma unroll
    for (int f = 0; f < 4; ++f)
#pragma unroll
      for (int e = 0; e < 8; ++e)
        Ss[(16 * wave + e + 8 * lh) * 65 + 16 * f + ln] = sacc[f][e] * 0.125f;
    __syncthreads();

    // online softmax (one thread per q row)
    if (tid < 64) {
      const int r = tid;
      const float mold = mrow[r];
      float rm = mold;
      for (int j = 0; j < 64; ++j) rm = fmaxf(rm, Ss[r * 65 + j]);
      const float alpha = __expf(mold - rm);
      float s = lrow[r] * alpha;
      for (int j = 0; j < 64; ++j) {
        const float p = __expf(Ss[r * 65 + j] - rm);
        s += p;
        Ps[r * 72 + j] = (bf16_t)p;
      }
      mrow[r] = rm; lrow[r] = s; arow[r] = alpha;
    }
    __syncthreads();

    // O = O*alpha + P @ V
    float al[8];
#pragma unroll
    for (int e = 0; e < 8; ++e) al[e] = arow[16 * wave + e + 8 * lh];
#pragma unroll
    for (int f = 0; f < 4; ++f)
#pragma unroll
      for (int e = 0; e < 8; ++e) oacc[f][e] *= al[e];
#pragma unroll
    for (int ks = 0; ks < 64; ks += 32) {
      v16bf ap = load_frag(Ps, 72, 16 * wave, ks, lane);
#pragma unroll
      for (int f = 0; f < 4; ++f) {
        v16bf bv = load_frag(Vt, 72, 16 * f, ks, lane);
        oacc[f] = wmma_bf16(ap, bv, oacc[f]);
      }
    }
    __syncthreads();
  }

  float li[8];
#pragma unroll
  for (int e = 0; e < 8; ++e) li[e] = 1.0f / lrow[16 * wave + e + 8 * lh];
#pragma unroll
  for (int f = 0; f < 4; ++f)
#pragma unroll
    for (int e = 0; e < 8; ++e) {
      const int row = 16 * wave + e + 8 * lh;
      o[(tok0 + row) * C_DIM + cq + 16 * f + ln] = (bf16_t)(oacc[f][e] * li[e]);
    }
}

// ---------------------------------------------------------------------------
// Launch
// ---------------------------------------------------------------------------
extern "C" void kernel_launch(void* const* d_in, const int* in_sizes, int n_in,
                              void* d_out, int out_size, void* d_ws, size_t ws_size,
                              hipStream_t stream) {
  (void)in_sizes; (void)n_in; (void)out_size; (void)ws_size;
  const float* feats  = (const float*)d_in[0];
  const float* mod    = (const float*)d_in[1];
  const float* w_mod  = (const float*)d_in[2];
  const float* b_mod  = (const float*)d_in[3];
  const float* w_qkv  = (const float*)d_in[4];
  const float* b_qkv  = (const float*)d_in[5];
  const float* w_proj = (const float*)d_in[6];
  const float* b_proj = (const float*)d_in[7];
  const float* w_fc1  = (const float*)d_in[8];
  const float* b_fc1  = (const float*)d_in[9];
  const float* w_fc2  = (const float*)d_in[10];
  const float* b_fc2  = (const float*)d_in[11];
  float* out = (float*)d_out;

  char* ws = (char*)d_ws;
  const size_t MB = 1u << 20;
  float*  mod6  = (float*)ws;                           // 48 KB (reserve 64K)
  bf16_t* hbuf  = (bf16_t*)(ws + (64u << 10));          //  8 MB [4096,1024]
  bf16_t* qkvb  = (bf16_t*)(ws + (64u << 10) +  8 * MB);// 24 MB [4096,3072]
  bf16_t* obuf  = (bf16_t*)(ws + (64u << 10) + 32 * MB);//  8 MB [4096,1024]
  bf16_t* ubuf  = (bf16_t*)(ws + (64u << 10) + 40 * MB);// 32 MB [4096,4096]
  bf16_t* wqkvT = (bf16_t*)(ws + (64u << 10) + 72 * MB);//  6 MB [3072,1024]
  bf16_t* wprojT= (bf16_t*)(ws + (64u << 10) + 78 * MB);//  2 MB [1024,1024]
  bf16_t* wfc1T = (bf16_t*)(ws + (64u << 10) + 80 * MB);//  8 MB [4096,1024]
  bf16_t* wfc2T = (bf16_t*)(ws + (64u << 10) + 88 * MB);//  8 MB [1024,4096]

  // Weight prep (bf16, transposed to [N,K]) + modulation vector
  transpose_cvt<<<dim3(32, 96), dim3(32, 8), 0, stream>>>(w_qkv, wqkvT, 1024, 3072);
  transpose_cvt<<<dim3(32, 32), dim3(32, 8), 0, stream>>>(w_proj, wprojT, 1024, 1024);
  transpose_cvt<<<dim3(32, 128), dim3(32, 8), 0, stream>>>(w_fc1, wfc1T, 1024, 4096);
  transpose_cvt<<<dim3(128, 32), dim3(32, 8), 0, stream>>>(w_fc2, wfc2T, 4096, 1024);
  mod_silu_gemm<<<dim3(24, 2), 256, 0, stream>>>(mod, w_mod, b_mod, mod6);

  // MSA branch
  ln_modulate<<<dim3(4096), 256, 0, stream>>>(feats, mod6, 0 * C_DIM, 1 * C_DIM, hbuf);
  gemm_wmma<EPI_BF16><<<dim3(48, 32), 128, 0, stream>>>(
      hbuf, wqkvT, b_qkv, qkvb, nullptr, nullptr, nullptr, 4096, 3072, 1024);
  flash_attn<<<dim3(32, 32), 128, 0, stream>>>(qkvb, obuf);
  gemm_wmma<EPI_RESID><<<dim3(16, 32), 128, 0, stream>>>(
      obuf, wprojT, b_proj, nullptr, out, feats, mod6 + 2 * C_DIM, 4096, 1024, 1024);

  // FFN branch
  ln_modulate<<<dim3(4096), 256, 0, stream>>>(out, mod6, 3 * C_DIM, 4 * C_DIM, hbuf);
  gemm_wmma<EPI_GELU><<<dim3(64, 32), 128, 0, stream>>>(
      hbuf, wfc1T, b_fc1, ubuf, nullptr, nullptr, nullptr, 4096, 4096, 1024);
  gemm_wmma<EPI_RESID><<<dim3(16, 32), 128, 0, stream>>>(
      ubuf, wfc2T, b_fc2, nullptr, out, out, mod6 + 5 * C_DIM, 4096, 1024, 4096);
}
